// EnhancedMoEGate_15281493639598
// MI455X (gfx1250) — compile-verified, matrix operand order checked by reading, not run on hardware
//
#include <hip/hip_runtime.h>
#include <hip/hip_bf16.h>
#include <math.h>

typedef __attribute__((ext_vector_type(16))) __bf16 v16bf;
typedef __attribute__((ext_vector_type(8)))  float  v8f;
typedef __attribute__((ext_vector_type(4)))  float  v4f;   // native vector for nontemporal loads

#define HIDDEN      4096
#define NEXP        64
#define TOPK        8
#define SOFTCAPV    30.0f
#define TOK_PER_BLK 32
#define LDS_STRIDE  65   // pad 64 -> 65 floats: conflict-free row reads
#define NEG_INF     (-3.0e38f)
#define NSTEP       ((HIDDEN / 4) / 32)   // 32 K-steps of 32 per K-quarter

union bfrag {
    v16bf    v;
    unsigned u[8];
    uint4    q[2];
};

// fp32 pair -> packed bf16 dword: round-half-up (+0x8000) then one v_perm_b32
__device__ __forceinline__ unsigned pk2u(float lo, float hi) {
    unsigned ul = __float_as_uint(lo) + 0x8000u;
    unsigned uh = __float_as_uint(hi) + 0x8000u;
    // result bytes: [uh.b3, uh.b2, ul.b3, ul.b2]  (src0=uh -> bytes 4..7)
    return __builtin_amdgcn_perm(uh, ul, 0x07060302u);
}

// ---- prep: gate_weight fp32 [E,H] -> bf16 [E,H] (row major, K contiguous) ----
__global__ void moe_gate_wcvt(const float* __restrict__ w,
                              unsigned* __restrict__ wb, int n2) {
    int i = blockIdx.x * blockDim.x + threadIdx.x;
    if (i < n2) {
        float2 f = ((const float2*)w)[i];
        wb[i] = pk2u(f.x, f.y);
    }
}

// ---- main: logits via v_wmma_f32_16x16x32_bf16, LDS reduce, top-8 ----
__global__ __launch_bounds__(256) void moe_gate_main(
        const float* __restrict__ x,
        const unsigned short* __restrict__ wb,   // bf16 bits [E][H]
        float* __restrict__ out, int T) {

    __shared__ float lds[2 * 16 * LDS_STRIDE];

    const int tid  = threadIdx.x;
    const int lane = tid & 31;
    const int wave = tid >> 5;
    const int tileIdx = wave & 1;     // which 16-token tile of this block
    const int kq      = wave >> 1;    // K quarter: 0..3
    const int blockTok = blockIdx.x * TOK_PER_BLK;

    // zero the reduction tile
    for (int i = tid; i < 2 * 16 * LDS_STRIDE; i += 256) lds[i] = 0.0f;
    __syncthreads();

    const int lmod = lane & 15;
    const int half = lane >> 4;

    int row = blockTok + tileIdx * 16 + lmod;
    if (row >= T) row = T - 1;                    // benign clamp
    const float* aptr = x + (size_t)row * HIDDEN;
    const unsigned short* bbase = wb + (size_t)lmod * HIDDEN;
    const int kbase = kq * (HIDDEN / 4);          // 1024-wide K slice

    v8f acc[4];
    #pragma unroll
    for (int t = 0; t < 4; ++t) acc[t] = (v8f){};

    // ---------- software pipeline: prefetch iteration 0 ----------
    v4f   a_cur[4];
    bfrag b_cur[4];
    {
        const int r0 = kbase + half * 8;
        const int r1 = r0 + 16;
        a_cur[0] = __builtin_nontemporal_load((const v4f*)(aptr + r0));
        a_cur[1] = __builtin_nontemporal_load((const v4f*)(aptr + r0) + 1);
        a_cur[2] = __builtin_nontemporal_load((const v4f*)(aptr + r1));
        a_cur[3] = __builtin_nontemporal_load((const v4f*)(aptr + r1) + 1);
        #pragma unroll
        for (int t = 0; t < 4; ++t) {
            b_cur[t].q[0] = *(const uint4*)(bbase + t * 16 * HIDDEN + r0);
            b_cur[t].q[1] = *(const uint4*)(bbase + t * 16 * HIDDEN + r1);
        }
    }

    #pragma unroll 2
    for (int kk = 0; kk < NSTEP; ++kk) {
        // ---- prefetch kk+1 (address clamped in-bounds; no branch) ----
        int k0n = kbase + (kk + 1) * 32;
        k0n = (k0n > HIDDEN - 32) ? (HIDDEN - 32) : k0n;
        const int n0 = k0n + half * 8;
        const int n1 = n0 + 16;

        v4f   a_nxt[4];
        bfrag b_nxt[4];
        a_nxt[0] = __builtin_nontemporal_load((const v4f*)(aptr + n0));
        a_nxt[1] = __builtin_nontemporal_load((const v4f*)(aptr + n0) + 1);
        a_nxt[2] = __builtin_nontemporal_load((const v4f*)(aptr + n1));
        a_nxt[3] = __builtin_nontemporal_load((const v4f*)(aptr + n1) + 1);
        #pragma unroll
        for (int t = 0; t < 4; ++t) {
            b_nxt[t].q[0] = *(const uint4*)(bbase + t * 16 * HIDDEN + n0);
            b_nxt[t].q[1] = *(const uint4*)(bbase + t * 16 * HIDDEN + n1);
        }

        // ---- convert current A: 8x (2 adds + 1 v_perm_b32) ----
        bfrag A;
        A.u[0] = pk2u(a_cur[0].x, a_cur[0].y);
        A.u[1] = pk2u(a_cur[0].z, a_cur[0].w);
        A.u[2] = pk2u(a_cur[1].x, a_cur[1].y);
        A.u[3] = pk2u(a_cur[1].z, a_cur[1].w);
        A.u[4] = pk2u(a_cur[2].x, a_cur[2].y);
        A.u[5] = pk2u(a_cur[2].z, a_cur[2].w);
        A.u[6] = pk2u(a_cur[3].x, a_cur[3].y);
        A.u[7] = pk2u(a_cur[3].z, a_cur[3].w);

        // ---- 4 WMMAs on data prefetched one iteration ago ----
        acc[0] = __builtin_amdgcn_wmma_f32_16x16x32_bf16(
                     false, A.v, false, b_cur[0].v, (short)0, acc[0], false, false);
        acc[1] = __builtin_amdgcn_wmma_f32_16x16x32_bf16(
                     false, A.v, false, b_cur[1].v, (short)0, acc[1], false, false);
        acc[2] = __builtin_amdgcn_wmma_f32_16x16x32_bf16(
                     false, A.v, false, b_cur[2].v, (short)0, acc[2], false, false);
        acc[3] = __builtin_amdgcn_wmma_f32_16x16x32_bf16(
                     false, A.v, false, b_cur[3].v, (short)0, acc[3], false, false);

        // ---- rotate pipeline registers ----
        #pragma unroll
        for (int j = 0; j < 4; ++j) a_cur[j] = a_nxt[j];
        #pragma unroll
        for (int t = 0; t < 4; ++t) b_cur[t] = b_nxt[t];
    }

    // reduce K-quarter partials into LDS (4 waves hit same tile -> ds_add_f32)
    float* tbase = lds + tileIdx * 16 * LDS_STRIDE;
    #pragma unroll
    for (int t = 0; t < 4; ++t) {
        const int col = t * 16 + lmod;
        #pragma unroll
        for (int r = 0; r < 8; ++r) {
            const int rr = r + half * 8;     // C layout: VGPR r = row r / r+8
            atomicAdd(&tbase[rr * LDS_STRIDE + col], acc[t][r]);
        }
    }
    __syncthreads();

    // post-processing: 32 lanes, one token each
    if (tid < 32) {
        const int token = blockTok + tid;
        if (token < T) {
            float* rowp = lds + (tid >> 4) * 16 * LDS_STRIDE
                              + (tid & 15) * LDS_STRIDE;
            float vals[TOPK]; int idxs[TOPK];
            #pragma unroll
            for (int p = 0; p < TOPK; ++p) {
                float m = NEG_INF; int mi = 0;
                for (int e = 0; e < NEXP; ++e) {
                    float v = rowp[e];
                    if (v > m) { m = v; mi = e; }
                }
                vals[p] = m; idxs[p] = mi;
                rowp[mi] = NEG_INF;
            }
            // tanh softcap only on winners (monotone -> same top-k)
            const float m0 = SOFTCAPV * tanhf(vals[0] / SOFTCAPV);
            float w8[TOPK]; float sum = 0.0f;
            #pragma unroll
            for (int p = 0; p < TOPK; ++p) {
                float sc = SOFTCAPV * tanhf(vals[p] / SOFTCAPV);
                float e  = __expf(sc - m0);
                w8[p] = e; sum += e;
            }
            const float inv = 1.0f / sum;
            float* rw = out + (size_t)token * TOPK;
            float* se = out + (size_t)T * TOPK + (size_t)token * TOPK;
            #pragma unroll
            for (int p = 0; p < TOPK; ++p) {
                rw[p] = w8[p] * inv;
                se[p] = (float)idxs[p];
            }
        }
    }
}

extern "C" void kernel_launch(void* const* d_in, const int* in_sizes, int n_in,
                              void* d_out, int out_size, void* d_ws, size_t ws_size,
                              hipStream_t stream) {
    const float* x  = (const float*)d_in[0];     // [B,S,H] fp32
    const float* gw = (const float*)d_in[1];     // [E,H]   fp32
    float* out = (float*)d_out;
    unsigned* wb = (unsigned*)d_ws;              // 512 KB bf16 gate weight

    const int T = in_sizes[0] / HIDDEN;          // 16384 tokens

    const int n2 = NEXP * HIDDEN / 2;
    moe_gate_wcvt<<<(n2 + 255) / 256, 256, 0, stream>>>(gw, wb, n2);

    const int nblk = (T + TOK_PER_BLK - 1) / TOK_PER_BLK;
    moe_gate_main<<<nblk, 256, 0, stream>>>(
        x, (const unsigned short*)wb, out, T);
}
